// SelfAttention_45097156608335
// MI455X (gfx1250) — matrix operand order repaired and checked by reading, not compiled, not run on hardware
//
#include <hip/hip_runtime.h>
#include <hip/hip_bf16.h>
#include <stdint.h>

// ---------------------------------------------------------------------------
// CDNA5 (gfx1250) self-attention block, fp16 WMMA with fp32 accumulation.
// GEMMs use async global->LDS copies (ASYNCcnt) with double-buffered tiles.
// ---------------------------------------------------------------------------

typedef __attribute__((ext_vector_type(16))) _Float16     v16h;
typedef __attribute__((ext_vector_type(8)))  float        v8f;
typedef __attribute__((ext_vector_type(4)))  unsigned int u32x4;

#define WMMA_F32_F16(A_, B_, C_) \
  __builtin_amdgcn_wmma_f32_16x16x32_f16(false, (A_), false, (B_), (short)0, (C_), false, false)

// Build a 16-half fragment from two contiguous 16-byte runs.
static __device__ __forceinline__ v16h ld2x16(const _Float16* p0, const _Float16* p1) {
  union { u32x4 q[2]; v16h v; } u;
  u.q[0] = *(const u32x4*)p0;
  u.q[1] = *(const u32x4*)p1;
  return u.v;
}

// Exchange value between lanes L and L^16 (wave32) via ds_swizzle SWAPX16.
static __device__ __forceinline__ float swz_xor16(float x) {
  return __int_as_float(__builtin_amdgcn_ds_swizzle(__float_as_int(x), 0x401F));
}

// CDNA5 async copy: 16B global -> LDS, tracked by ASYNCcnt (no VGPR round-trip).
static __device__ __forceinline__ void async_cp16(const _Float16* g, _Float16* l) {
  unsigned lds = (unsigned)(uintptr_t)l;          // low 32 bits = LDS byte offset
  unsigned long long ga = (unsigned long long)(uintptr_t)g;
  asm volatile("global_load_async_to_lds_b128 %0, %1, off"
               :: "v"(lds), "v"(ga)
               : "memory");
}

static __device__ __forceinline__ void wait_async0() {
  asm volatile("s_wait_asynccnt 0x0" ::: "memory");
}

// ---------------------------------------------------------------------------
// Generic fp32 -> fp16 conversion (n multiple of 4).
// ---------------------------------------------------------------------------
__global__ __launch_bounds__(256) void cvt_f32_to_f16(const float* __restrict__ in,
                                                      _Float16* __restrict__ out, int n) {
  int i = (blockIdx.x * 256 + threadIdx.x) * 4;
  if (i < n) {
    out[i + 0] = (_Float16)in[i + 0];
    out[i + 1] = (_Float16)in[i + 1];
    out[i + 2] = (_Float16)in[i + 2];
    out[i + 3] = (_Float16)in[i + 3];
  }
}

// ---------------------------------------------------------------------------
// Tiled NT GEMM:  C[M,N] (f32) = A[M,K] (f16, row-major) * B[N,K]^T (f16, row-major)
// Both operands K-contiguous -> WMMA A/B fragments are contiguous LDS loads.
// Block tile 128x64, BK=64 (two WMMA k-steps per staged tile), 8 waves in 4x2,
// each wave a 32x32 tile. LDS tiles double-buffered, filled with async copies.
// ---------------------------------------------------------------------------
__global__ __launch_bounds__(256) void gemm_nt_f16f32(
    const _Float16* __restrict__ A, const _Float16* __restrict__ B,
    float* __restrict__ C, const float* __restrict__ bias,
    int M, int N, int K) {
  constexpr int BM = 128, BN = 64, BK = 64, LDT = 72;  // 72 halves = 144B padded stride
  __shared__ _Float16 lA[2][BM * LDT];
  __shared__ _Float16 lB[2][BN * LDT];

  const int tid  = threadIdx.x;
  const int lane = tid & 31, ln = lane & 15, hi = lane >> 4;
  const int wv   = tid >> 5;
  const int bm = blockIdx.y * BM, bn = blockIdx.x * BN;
  const int wr = (wv >> 1) * 32, wc = (wv & 1) * 32;

  // Stage one BK-wide tile pair into LDS buffer `buf` (async, 16B granules).
  auto stage = [&](int k0, int buf) {
    #pragma unroll
    for (int v = 0; v < 4; ++v) {            // A: 128x64 halves = 1024 x 16B
      int idx = tid + v * 256;
      int row = idx >> 3, ko = (idx & 7) * 8;
      async_cp16(A + (size_t)(bm + row) * K + k0 + ko, &lA[buf][row * LDT + ko]);
    }
    #pragma unroll
    for (int v = 0; v < 2; ++v) {            // B: 64x64 halves = 512 x 16B
      int idx = tid + v * 256;
      int row = idx >> 3, ko = (idx & 7) * 8;
      async_cp16(B + (size_t)(bn + row) * K + k0 + ko, &lB[buf][row * LDT + ko]);
    }
  };

  v8f acc[2][2] = {};

  stage(0, 0);
  wait_async0();
  __syncthreads();

  int buf = 0;
  for (int k0 = 0; k0 < K; k0 += BK, buf ^= 1) {
    if (k0 + BK < K) stage(k0 + BK, buf ^ 1);   // prefetch next tile into other buffer

    #pragma unroll
    for (int ks = 0; ks < 2; ++ks) {            // two 16x16x32 k-steps per tile
      v16h af[2], bf[2];
      #pragma unroll
      for (int i = 0; i < 2; ++i) {
        // A fragment (16x32): lane ln = row; elems 0..7 -> k = hi*8+0..7, 8..15 -> +16
        const _Float16* p = &lA[buf][(wr + i * 16 + ln) * LDT + ks * 32 + hi * 8];
        af[i] = ld2x16(p, p + 16);
      }
      #pragma unroll
      for (int j = 0; j < 2; ++j) {
        // B fragment (32x16): lane ln = col; elems 0..7 -> k = hi*8+0..7, 8..15 -> +16
        const _Float16* p = &lB[buf][(wc + j * 16 + ln) * LDT + ks * 32 + hi * 8];
        bf[j] = ld2x16(p, p + 16);
      }
      #pragma unroll
      for (int i = 0; i < 2; ++i)
        #pragma unroll
        for (int j = 0; j < 2; ++j)
          acc[i][j] = WMMA_F32_F16(af[i], bf[j], acc[i][j]);
    }

    wait_async0();       // our async writes into buf^1 are in LDS
    __syncthreads();     // everyone done reading buf / writing buf^1
  }

  // Epilogue: C layout — lane ln = column, vgpr r = row (r, or r+8 for upper lanes).
  #pragma unroll
  for (int i = 0; i < 2; ++i) {
    #pragma unroll
    for (int j = 0; j < 2; ++j) {
      int n = bn + wc + j * 16 + ln;
      float bv = bias ? bias[n] : 0.0f;
      #pragma unroll
      for (int r = 0; r < 8; ++r) {
        int m = bm + wr + i * 16 + r + hi * 8;
        C[(size_t)m * N + n] = acc[i][j][r] + bv;
      }
    }
  }
}

// ---------------------------------------------------------------------------
// RMSNorm + rotary for q,k; convert + transpose v -> vt[H][D][T].
// One block per (t, h), 128 threads = one per d.
// ---------------------------------------------------------------------------
__global__ __launch_bounds__(128) void qkv_post(
    const float* __restrict__ QKV, _Float16* __restrict__ qh,
    _Float16* __restrict__ kh, _Float16* __restrict__ vt) {
  const int t = blockIdx.x, h = blockIdx.y, d = threadIdx.x;
  __shared__ float red[128];
  __shared__ float buf[128];
  const float* row = QKV + (size_t)t * 6144;

  const int   i   = d & 63;                                   // rotary pair index
  const float ang = (float)t * __expf((float)i * -0.14391156831212787f); // 10000^(-i/64)
  float sn, cs;
  __sincosf(ang, &sn, &cs);

  #pragma unroll
  for (int part = 0; part < 2; ++part) {   // 0 = q, 1 = k
    _Float16* dst = part ? kh : qh;
    float xv = row[part * 2048 + h * 128 + d];
    red[d] = xv * xv;
    __syncthreads();
    #pragma unroll
    for (int s2 = 64; s2 > 0; s2 >>= 1) {
      if (d < s2) red[d] += red[d + s2];
      __syncthreads();
    }
    float yv = xv * rsqrtf(red[0] * (1.0f / 128.0f) + 1.1920929e-7f);
    buf[d] = yv;
    __syncthreads();
    float x1 = buf[i], x2 = buf[i + 64];
    float r = (d < 64) ? (x1 * cs + x2 * sn) : (x2 * cs - x1 * sn);
    dst[((size_t)t * 16 + h) * 128 + d] = (_Float16)r;
    __syncthreads();
  }

  float vv = row[4096 + h * 128 + d];
  vt[((size_t)(h * 128 + d)) * 2048 + t] = (_Float16)vv;   // transposed for PV GEMM
}

// ---------------------------------------------------------------------------
// Attention. One wave handles (head h, 16-query tile). Scores computed
// transposed: St = K_chunk * Q^T so each lane owns one query column of St —
// softmax statistics are per-lane scalars (combined lane<->lane+16 via
// ds_swizzle). St's C-layout re-packs directly into P's A-fragment layout.
// ---------------------------------------------------------------------------
static __device__ __forceinline__ void score_tiles(
    const _Float16* __restrict__ kh, int h, int kc, int ln, int hi,
    const v16h qf[4], v8f& st0, v8f& st1) {
  v8f s0 = {};
  v8f s1 = {};
  #pragma unroll
  for (int ds = 0; ds < 4; ++ds) {
    const _Float16* kp0 = kh + ((size_t)(kc + ln) * 16 + h) * 128 + ds * 32 + hi * 8;
    v16h ka0 = ld2x16(kp0, kp0 + 16);
    s0 = WMMA_F32_F16(ka0, qf[ds], s0);
    const _Float16* kp1 = kh + ((size_t)(kc + 16 + ln) * 16 + h) * 128 + ds * 32 + hi * 8;
    v16h ka1 = ld2x16(kp1, kp1 + 16);
    s1 = WMMA_F32_F16(ka1, qf[ds], s1);
  }
  st0 = s0;
  st1 = s1;
}

__global__ __launch_bounds__(128) void attn_fwd(
    const _Float16* __restrict__ qh, const _Float16* __restrict__ kh,
    const _Float16* __restrict__ vt, float* __restrict__ y) {
  const int lane = threadIdx.x & 31, ln = lane & 15, hi = lane >> 4;
  const int wg = (blockIdx.x * blockDim.x + threadIdx.x) >> 5;
  const int h = wg >> 7, qt = wg & 127;
  const int qbase = qt * 16;
  const int qg = qbase + ln;        // this lane's query (column of St)
  const float scale = 0.08838834764831845f;   // 1/sqrt(128)

  // Resident Q B-fragments: lane ln = query col, elems 0..7 -> d = hi*8+0..7, 8..15 -> +16.
  v16h qf[4];
  {
    const _Float16* qp = qh + ((size_t)qg * 16 + h) * 128;
    #pragma unroll
    for (int ds = 0; ds < 4; ++ds) {
      const _Float16* p = qp + ds * 32 + hi * 8;
      qf[ds] = ld2x16(p, p + 16);
    }
  }

  const int nch = (qbase + 47) >> 5;   // causal: only key chunks <= qbase+15

  // Pass 1: online max / sum-of-exp per query (per-lane scalars).
  float mx = -1e30f, sm = 0.0f;
  for (int c = 0; c < nch; ++c) {
    const int kc = c << 5;
    v8f st0, st1;
    score_tiles(kh, h, kc, ln, hi, qf, st0, st1);
    #pragma unroll
    for (int r = 0; r < 8; ++r) {
      int k0 = kc + r + hi * 8;
      float v0 = (k0 <= qg) ? st0[r] * scale : -1e30f;
      float n0 = fmaxf(mx, v0);
      sm = sm * __expf(mx - n0) + ((k0 <= qg) ? __expf(v0 - n0) : 0.0f);
      mx = n0;
      int k1 = k0 + 16;
      float v1 = (k1 <= qg) ? st1[r] * scale : -1e30f;
      float n1 = fmaxf(mx, v1);
      sm = sm * __expf(mx - n1) + ((k1 <= qg) ? __expf(v1 - n1) : 0.0f);
      mx = n1;
    }
  }
  {  // combine the two lane halves that own the same query column
    float om = swz_xor16(mx), os = swz_xor16(sm);
    float nm = fmaxf(mx, om);
    sm = sm * __expf(mx - nm) + os * __expf(om - nm);
    mx = nm;
  }
  const float rden = 1.0f / sm;

  // Pass 2: recompute scores, P = exp(S - mx)/sum, PV accumulate (D = 8 tiles).
  v8f acc[8] = {};
  for (int c = 0; c < nch; ++c) {
    const int kc = c << 5;
    v8f st0, st1;
    score_tiles(kh, h, kc, ln, hi, qf, st0, st1);
    // St C-layout repacks directly into P's A-fragment layout (lane = query row).
    v16h pf;
    #pragma unroll
    for (int r = 0; r < 8; ++r) {
      int k0 = kc + r + hi * 8;
      float p0 = (k0 <= qg) ? __expf(st0[r] * scale - mx) * rden : 0.0f;
      pf[r] = (_Float16)p0;
      int k1 = k0 + 16;
      float p1 = (k1 <= qg) ? __expf(st1[r] * scale - mx) * rden : 0.0f;
      pf[r + 8] = (_Float16)p1;
    }
    #pragma unroll
    for (int dt = 0; dt < 8; ++dt) {
      // V B-fragment from transposed V: lane ln = d col, keys hi*8+0..7 and +16.
      const _Float16* vp = vt + ((size_t)(h * 128 + dt * 16 + ln)) * 2048 + kc + hi * 8;
      v16h vf = ld2x16(vp, vp + 16);
      acc[dt] = WMMA_F32_F16(pf, vf, acc[dt]);
    }
  }

  // Store y[T, Hs] (f32): C layout, column = d, row = query.
  #pragma unroll
  for (int dt = 0; dt < 8; ++dt) {
    int col = h * 128 + dt * 16 + ln;
    #pragma unroll
    for (int r = 0; r < 8; ++r) {
      int m = qbase + r + hi * 8;
      y[(size_t)m * 2048 + col] = acc[dt][r];
    }
  }
}

// ---------------------------------------------------------------------------
// Host-side orchestration.
// ---------------------------------------------------------------------------
extern "C" void kernel_launch(void* const* d_in, const int* in_sizes, int n_in,
                              void* d_out, int out_size, void* d_ws, size_t ws_size,
                              hipStream_t stream) {
  (void)in_sizes; (void)n_in; (void)out_size; (void)ws_size;
  const float* x   = (const float*)d_in[0];   // [2048, 2048]
  const float* Wq  = (const float*)d_in[1];   // [3*2048, 2048] flattened
  const float* Wo  = (const float*)d_in[2];   // [2048, 2048]
  const float* Wob = (const float*)d_in[3];   // [2048]
  float* out = (float*)d_out;                 // [2048, 2048]

  char* w = (char*)d_ws;
  auto take = [&](size_t bytes) -> char* {
    char* p = w;
    w += (bytes + 255) & ~(size_t)255;
    return p;
  };
  _Float16* Xh  = (_Float16*)take(2048ull * 2048 * 2);   //  8 MB x fp16
  _Float16* Wh  = (_Float16*)take(6144ull * 2048 * 2);   // 24 MB Wqkv fp16
  _Float16* Woh = (_Float16*)take(2048ull * 2048 * 2);   //  8 MB Wo fp16
  float*    QKV = (float*)   take(2048ull * 6144 * 4);   // 48 MB qkv fp32
  _Float16* qh  = (_Float16*)take(2048ull * 2048 * 2);   //  8 MB q (norm+rope) fp16
  _Float16* kh  = (_Float16*)take(2048ull * 2048 * 2);   //  8 MB k
  _Float16* vtb = (_Float16*)take(2048ull * 2048 * 2);   //  8 MB v transposed
  float*    yf  = (float*)   take(2048ull * 2048 * 4);   // 16 MB attn out fp32
  _Float16* yh  = (_Float16*)take(2048ull * 2048 * 2);   //  8 MB attn out fp16

  // 1) fp32 -> fp16 operand conversion.
  cvt_f32_to_f16<<<dim3(2048 * 2048 / 4 / 256), dim3(256), 0, stream>>>(x,  Xh,  2048 * 2048);
  cvt_f32_to_f16<<<dim3(6144 * 2048 / 4 / 256), dim3(256), 0, stream>>>(Wq, Wh,  6144 * 2048);
  cvt_f32_to_f16<<<dim3(2048 * 2048 / 4 / 256), dim3(256), 0, stream>>>(Wo, Woh, 2048 * 2048);

  // 2) QKV = X @ Wqkv^T   (M=2048, N=6144, K=2048)
  gemm_nt_f16f32<<<dim3(6144 / 64, 2048 / 128), dim3(256), 0, stream>>>(
      Xh, Wh, QKV, nullptr, 2048, 6144, 2048);

  // 3) RMSNorm + rotary (q,k); transpose v.
  qkv_post<<<dim3(2048, 16), dim3(128), 0, stream>>>(QKV, qh, kh, vtb);

  // 4) Causal attention (2048 waves: 16 heads x 128 query tiles).
  attn_fwd<<<dim3(512), dim3(128), 0, stream>>>(qh, kh, vtb, yf);

  // 5) Output projection: out = y @ Wo^T + b   (M=N=K=2048)
  cvt_f32_to_f16<<<dim3(2048 * 2048 / 4 / 256), dim3(256), 0, stream>>>(yf, yh, 2048 * 2048);
  gemm_nt_f16f32<<<dim3(2048 / 64, 2048 / 128), dim3(256), 0, stream>>>(
      yh, Woh, out, Wob, 2048, 2048, 2048);
}
